// TSRL_59442347377404
// MI455X (gfx1250) — matrix operand order repaired and checked by reading, not compiled, not run on hardware
//
#include <hip/hip_runtime.h>
#include <hip/hip_bf16.h>
#include <math.h>

typedef __bf16 bf16_t;
typedef __attribute__((ext_vector_type(16))) __bf16 v16bf;
typedef __attribute__((ext_vector_type(8)))  __bf16 v8bf;
typedef __attribute__((ext_vector_type(8)))  float  v8f;

union ABf16 { v16bf v; v8bf h[2]; };

__device__ __forceinline__ v8f zero8() {
  v8f z;
#pragma unroll
  for (int i = 0; i < 8; ++i) z[i] = 0.0f;
  return z;
}

__device__ __forceinline__ v16bf zero16bf() {
  v16bf z;
#pragma unroll
  for (int i = 0; i < 16; ++i) z[i] = (bf16_t)0.0f;
  return z;
}

__device__ __forceinline__ v8f wmma_bf16(v16bf a, v16bf b, v8f c) {
  // D = A(16x32 bf16) * B(32x16 bf16) + C(16x16 f32)
  return __builtin_amdgcn_wmma_f32_16x16x32_bf16(false, a, false, b, (short)0, c,
                                                 false, false);
}

__device__ __forceinline__ float gelu_f(float x) {
  return 0.5f * x * (1.0f + erff(x * 0.70710678118654752f));
}

template<int ACT>
__device__ __forceinline__ float act_f(float s) {
  if (ACT == 1) return gelu_f(s);
  if (ACT == 2) return fmaxf(s, 0.0f);
  return s;
}

// 16-bit A (16x32), this code's contiguous k order (not the ISA table order): lane l: m = l&15, h = l>>4,
// holds K = [16h .. 16h+15]  (two 16B contiguous chunks, matching B)
__device__ __forceinline__ v16bf load_afrag(const bf16_t* __restrict__ row, int kk, int half) {
  ABf16 u;
  u.h[0] = *(const v8bf*)(row + kk + 16 * half);
  u.h[1] = *(const v8bf*)(row + kk + 16 * half + 8);
  return u.v;
}

// ---------------------------------------------------------------------------
// Generic WMMA GEMM:  out = ACT( A[M,K](bf16) @ W[N,K](bf16)^T + bias (+res) )
// Each wave computes a 16 x (16*NWT) tile. Block = 256 threads = 8 waves.
// PERM: store row' = (m % permQ) * permP + m / permQ  (layout permutation).
// ---------------------------------------------------------------------------
template<int NWT, int ACT, bool BIAS, bool RES, bool PERM>
__global__ __launch_bounds__(256)
void wgemm_kernel(const bf16_t* __restrict__ A, const bf16_t* __restrict__ W,
                  const float* __restrict__ bias, const float* __restrict__ res,
                  float* __restrict__ outF, bf16_t* __restrict__ outB,
                  bf16_t* __restrict__ outB2,
                  int M, int N, int K, int Nout, int permQ, int permP)
{
  const int lane   = threadIdx.x & 31;
  const int wid    = blockIdx.x * (blockDim.x >> 5) + (threadIdx.x >> 5);
  const int mtiles = M >> 4;
  const int ntw    = N / (16 * NWT);
  if (wid >= mtiles * ntw) return;
  const int mi   = wid % mtiles;
  const int n0   = (wid / mtiles) * (16 * NWT);
  const int half = lane >> 4, l15 = lane & 15;

  const bf16_t* Arow = A + (size_t)(mi * 16 + l15) * K;

  v8f acc[NWT];
#pragma unroll
  for (int j = 0; j < NWT; ++j) acc[j] = zero8();

  for (int kk = 0; kk < K; kk += 32) {
    __builtin_prefetch(Arow + kk + 64, 0, 1);          // global_prefetch_b8
    const v16bf a = load_afrag(Arow, kk, half);
#pragma unroll
    for (int j = 0; j < NWT; ++j) {
      // B (32x16): lane l: n = l&15, h = l>>4, K = [16h .. 16h+15] contiguous
      const v16bf b = *(const v16bf*)(W + (size_t)(n0 + j * 16 + l15) * K + kk + 16 * half);
      acc[j] = wmma_bf16(a, b, acc[j]);
    }
  }

#pragma unroll
  for (int j = 0; j < NWT; ++j) {
    const int n = n0 + j * 16 + l15;
    if (n >= Nout) continue;
    const float bv = BIAS ? bias[n] : 0.0f;
#pragma unroll
    for (int r = 0; r < 8; ++r) {
      const int m = mi * 16 + r + 8 * half;            // C/D: m = r + 8*(lane>>4)
      float s = acc[j][r] + bv;
      if (RES) s += res[(size_t)m * Nout + n];
      const size_t orow = PERM ? ((size_t)(m % permQ) * permP + (m / permQ)) : (size_t)m;
      const float v = act_f<ACT>(s);
      if (outF)  outF[orow * Nout + n]  = v;
      if (outB)  outB[orow * Nout + n]  = (bf16_t)v;
      if (outB2) outB2[orow * Nout + n] = (bf16_t)gelu_f(s);
    }
  }
}

// ---------------------------------------------------------------------------
// Dilated conv1d (k=3, 'same') as 3 shifted WMMA GEMMs over [B*T, K] rows.
// W3 = bf16 [3][N][K].  Boundary rows contribute zero A-fragments.
// ---------------------------------------------------------------------------
template<int ACT, bool RES>
__global__ __launch_bounds__(256)
void wconv3_kernel(const bf16_t* __restrict__ A, const bf16_t* __restrict__ W3,
                   const float* __restrict__ bias, const float* __restrict__ res,
                   float* __restrict__ outF, bf16_t* __restrict__ outB,
                   bf16_t* __restrict__ outB2,
                   int M, int N, int K, int T_, int dil)
{
  const int NWT    = 4;
  const int lane   = threadIdx.x & 31;
  const int wid    = blockIdx.x * (blockDim.x >> 5) + (threadIdx.x >> 5);
  const int mtiles = M >> 4;
  const int ntw    = N / 64;
  if (wid >= mtiles * ntw) return;
  const int mi   = wid % mtiles;
  const int n0   = (wid / mtiles) * 64;
  const int half = lane >> 4, l15 = lane & 15;

  const int row = mi * 16 + l15;
  const int bb = row / T_, tt0 = row % T_;

  v8f acc[NWT];
#pragma unroll
  for (int j = 0; j < NWT; ++j) acc[j] = zero8();

  for (int tap = 0; tap < 3; ++tap) {
    const int t2 = tt0 + (tap - 1) * dil;
    const bool ok = (t2 >= 0) && (t2 < T_);
    const bf16_t* Arow = A + (size_t)(bb * T_ + (ok ? t2 : tt0)) * K;
    const bf16_t* Wt = W3 + (size_t)tap * N * K;
    for (int kk = 0; kk < K; kk += 32) {
      v16bf a;
      if (ok) a = load_afrag(Arow, kk, half);
      else    a = zero16bf();                           // zero-pad halo (EXEC stays all-1s)
#pragma unroll
      for (int j = 0; j < NWT; ++j) {
        const v16bf b = *(const v16bf*)(Wt + (size_t)(n0 + j * 16 + l15) * K + kk + 16 * half);
        acc[j] = wmma_bf16(a, b, acc[j]);
      }
    }
  }

#pragma unroll
  for (int j = 0; j < NWT; ++j) {
    const int n = n0 + j * 16 + l15;
    const float bv = bias[n];
#pragma unroll
    for (int r = 0; r < 8; ++r) {
      const int m = mi * 16 + r + 8 * half;
      float s = acc[j][r] + bv;
      if (RES) s += res[(size_t)m * N + n];
      const float v = act_f<ACT>(s);
      if (outF)  outF[(size_t)m * N + n]  = v;
      if (outB)  outB[(size_t)m * N + n]  = (bf16_t)v;
      if (outB2) outB2[(size_t)m * N + n] = (bf16_t)gelu_f(s);
    }
  }
}

// ---------------------------------------------------------------------------
// GRU: 512 sequential steps. Each block owns 16 batch rows of one chain.
// h kept in LDS (f32 master + bf16 copy); gh = h@Whh^T via WMMA from LDS;
// gate math elementwise; o = h@Wout via WMMA; mask/y1 override fused.
// ---------------------------------------------------------------------------
__global__ __launch_bounds__(128)
void gru_kernel(const float* __restrict__ gi1, const float* __restrict__ gi2,
                const float* __restrict__ y1, const float* __restrict__ mask,
                const bf16_t* __restrict__ Whh, const float* __restrict__ bhh,
                const bf16_t* __restrict__ Wout, const float* __restrict__ bout,
                bf16_t* __restrict__ o1, bf16_t* __restrict__ o2, int T_, int Bb)
{
  const float* gi = blockIdx.y ? gi2 : gi1;
  bf16_t* oo = blockIdx.y ? o2 : o1;
  const int row0 = blockIdx.x * 16;

  __shared__ bf16_t sWhh[192 * 64];   // 24 KB
  __shared__ float  sGh[16 * 192];    // 12 KB
  __shared__ float  sHf[16 * 64];     //  4 KB
  __shared__ bf16_t sHb[16 * 64];     //  2 KB

  for (int i = threadIdx.x; i < 192 * 64; i += blockDim.x) sWhh[i] = Whh[i];
  for (int i = threadIdx.x; i < 16 * 64; i += blockDim.x) { sHf[i] = 0.0f; sHb[i] = (bf16_t)0.0f; }
  __syncthreads();

  const int lane = threadIdx.x & 31, wv = threadIdx.x >> 5;
  const int half = lane >> 4, l15 = lane & 15;

  for (int t = 0; t < T_; ++t) {
    // gh = h @ Whh^T + bhh : 12 n-tiles, 3 per wave
    for (int jj = 0; jj < 3; ++jj) {
      const int n0 = (wv * 3 + jj) * 16;
      v8f acc = zero8();
#pragma unroll
      for (int kk = 0; kk < 64; kk += 32) {
        const v16bf a = load_afrag(sHb + l15 * 64, kk, half);
        const v16bf b = *(const v16bf*)(sWhh + (n0 + l15) * 64 + kk + 16 * half);
        acc = wmma_bf16(a, b, acc);
      }
#pragma unroll
      for (int r = 0; r < 8; ++r) {
        const int m = r + 8 * half, n = n0 + l15;
        sGh[m * 192 + n] = acc[r] + bhh[n];
      }
    }
    __syncthreads();

    // gate update (torch GRUCell math)
    for (int e = threadIdx.x; e < 16 * 64; e += blockDim.x) {
      const int m = e >> 6, j = e & 63;
      const size_t gb = ((size_t)t * Bb + row0 + m) * 192 + j;
      const float ir = gi[gb], iz = gi[gb + 64], inn = gi[gb + 128];
      const float hr = sGh[m * 192 + j], hz = sGh[m * 192 + j + 64], hn = sGh[m * 192 + j + 128];
      const float rg = 1.0f / (1.0f + __expf(-(ir + hr)));
      const float zg = 1.0f / (1.0f + __expf(-(iz + hz)));
      const float ng = tanhf(inn + rg * hn);
      const float h = (1.0f - zg) * ng + zg * sHf[e];
      sHf[e] = h;
      sHb[e] = (bf16_t)h;
    }
    __syncthreads();

    // o = h @ out_w + out_b, masked override with y1_t : 20 n-tiles, 5 per wave
    for (int jj = 0; jj < 5; ++jj) {
      const int n0 = (wv * 5 + jj) * 16;
      v8f acc = zero8();
#pragma unroll
      for (int kk = 0; kk < 64; kk += 32) {
        const v16bf a = load_afrag(sHb + l15 * 64, kk, half);
        const v16bf b = *(const v16bf*)(Wout + (size_t)(n0 + l15) * 64 + kk + 16 * half);
        acc = wmma_bf16(a, b, acc);
      }
#pragma unroll
      for (int r = 0; r < 8; ++r) {
        const int m = r + 8 * half, n = n0 + l15;
        const int b_ = row0 + m;
        float o = acc[r] + bout[n];
        const bool keep = (mask[(size_t)b_ * T_ + t] == 1.0f);
        const float val = keep ? y1[((size_t)b_ * T_ + t) * 320 + n] : o;
        oo[((size_t)t * Bb + b_) * 320 + n] = (bf16_t)val;
      }
    }
    __syncthreads();
  }
}

// ---------------------------------------------------------------------------
// Small VALU / elementwise kernels (K=12 paths not worth WMMA)
// ---------------------------------------------------------------------------
__global__ __launch_bounds__(256)
void proj_gelu_kernel(const float* __restrict__ x, const float* __restrict__ w,
                      const float* __restrict__ b, float* __restrict__ outF,
                      bf16_t* __restrict__ outGB, int M)
{
  const int idx = blockIdx.x * blockDim.x + threadIdx.x;
  if (idx >= M * 64) return;
  const int row = idx >> 6, n = idx & 63;
  const float* xr = x + (size_t)row * 12;
  float s = b[n];
#pragma unroll
  for (int k = 0; k < 12; ++k) s += xr[k] * w[k * 64 + n];
  outF[idx] = s;
  outGB[idx] = (bf16_t)gelu_f(s);
}

__global__ __launch_bounds__(256)
void fc_kernel(const float* __restrict__ xi, const float* __restrict__ w,
               const float* __restrict__ b, float* __restrict__ outF,
               bf16_t* __restrict__ outB, int M)
{
  const int idx = blockIdx.x * blockDim.x + threadIdx.x;
  if (idx >= M * 320) return;
  const int row = idx / 320, n = idx % 320;
  const float* xr = xi + (size_t)row * 12;
  float s = b[n];
#pragma unroll
  for (int k = 0; k < 12; ++k) s += xr[k] * w[k * 320 + n];
  if (outF) outF[idx] = s;
  if (outB) outB[idx] = (bf16_t)s;
}

__global__ __launch_bounds__(256)
void mask_kernel(const float* __restrict__ xi, float* __restrict__ mask, int M)
{
  const int r = blockIdx.x * blockDim.x + threadIdx.x;
  if (r >= M) return;
  const float* p = xi + (size_t)r * 12;
  bool nan = false;
#pragma unroll
  for (int k = 0; k < 12; ++k) nan |= (p[k] != p[k]);
  mask[r] = nan ? 0.0f : 1.0f;
}

__global__ __launch_bounds__(256)
void misc_out_kernel(const float* __restrict__ x1, const float* __restrict__ x2,
                     float* __restrict__ m1, float* __restrict__ m2,
                     float* __restrict__ c1, float* __restrict__ c2, int n)
{
  const int idx = blockIdx.x * blockDim.x + threadIdx.x;
  if (idx >= n) return;
  const float a = x1[idx], b = x2[idx];
  m1[idx] = (a != 0.0f) ? 1.0f : 0.0f;
  m2[idx] = (b != 0.0f) ? 1.0f : 0.0f;
  c1[idx] = a;
  c2[idx] = b;
}

// Weight packing: f32 -> bf16 canonical [N,K]; transp: src is [K,Nsrc]; rows >= Nsrc zero.
__global__ __launch_bounds__(256)
void pack_nk_kernel(const float* __restrict__ src, bf16_t* __restrict__ dst,
                    int N, int K, int Nsrc, int transp)
{
  const int idx = blockIdx.x * blockDim.x + threadIdx.x;
  if (idx >= N * K) return;
  const int n = idx / K, k = idx % K;
  float v = 0.0f;
  if (n < Nsrc) v = transp ? src[(size_t)k * Nsrc + n] : src[(size_t)n * K + k];
  dst[idx] = (bf16_t)v;
}

// Conv weights [O,I,3] -> bf16 [3][O][I]
__global__ __launch_bounds__(256)
void pack_conv_kernel(const float* __restrict__ src, bf16_t* __restrict__ dst, int O, int I)
{
  const int idx = blockIdx.x * blockDim.x + threadIdx.x;
  if (idx >= 3 * O * I) return;
  const int k = idx / (O * I), r = idx % (O * I);
  const int o = r / I, i = r % I;
  dst[idx] = (bf16_t)src[((size_t)o * I + i) * 3 + k];
}

// ---------------------------------------------------------------------------
extern "C" void kernel_launch(void* const* d_in, const int* in_sizes, int n_in,
                              void* d_out, int out_size, void* d_ws, size_t ws_size,
                              hipStream_t stream)
{
  (void)in_sizes; (void)n_in; (void)out_size; (void)ws_size;
  const int Bc = 128, Tc = 512, BT = Bc * Tc;

  const float* x1    = (const float*)d_in[0];
  const float* x2    = (const float*)d_in[1];
  const float* X1    = (const float*)d_in[2];
  const float* X2    = (const float*)d_in[3];
  const float* w_in  = (const float*)d_in[5];
  const float* b_in  = (const float*)d_in[6];
  const float* c01w  = (const float*)d_in[7];
  const float* c01b  = (const float*)d_in[8];
  const float* c02w  = (const float*)d_in[9];
  const float* c02b  = (const float*)d_in[10];
  const float* c1pw  = (const float*)d_in[11];
  const float* c1pb  = (const float*)d_in[12];
  const float* c11w  = (const float*)d_in[13];
  const float* c11b  = (const float*)d_in[14];
  const float* c12w  = (const float*)d_in[15];
  const float* c12b  = (const float*)d_in[16];
  const float* ihdw  = (const float*)d_in[17];
  const float* ihdb  = (const float*)d_in[18];
  const float* ihpw  = (const float*)d_in[19];
  const float* ihpb  = (const float*)d_in[20];
  const float* fcw   = (const float*)d_in[21];
  const float* fcb   = (const float*)d_in[22];
  const float* wih   = (const float*)d_in[23];
  const float* whh   = (const float*)d_in[24];
  const float* bih   = (const float*)d_in[25];
  const float* bhh   = (const float*)d_in[26];
  const float* outw  = (const float*)d_in[27];
  const float* outb  = (const float*)d_in[28];
  const float* fcrw  = (const float*)d_in[29];
  const float* fcrb  = (const float*)d_in[30];

  float* out = (float*)d_out;
  const size_t OFF_XW2 = (size_t)BT * 320;
  const size_t OFF_O1  = 2 * OFF_XW2;
  const size_t OFF_O2  = OFF_O1 + (size_t)BT * 12;
  const size_t OFF_M1  = OFF_O2 + (size_t)BT * 12;
  const size_t OFF_M2  = OFF_M1 + (size_t)BT * 12;
  const size_t OFF_X1C = OFF_M2 + (size_t)BT * 12;
  const size_t OFF_X2C = OFF_X1C + (size_t)BT * 12;

  // ---- workspace carve (deterministic bump allocator) ----
  char* wsp = (char*)d_ws;
  size_t off = 0;
  auto carve = [&](size_t bytes) -> char* {
    char* p = wsp + off;
    off = (off + bytes + 255) & ~(size_t)255;
    return p;
  };
  bf16_t* pk_c01 = (bf16_t*)carve((size_t)3 * 64 * 64 * 2);
  bf16_t* pk_c02 = (bf16_t*)carve((size_t)3 * 64 * 64 * 2);
  bf16_t* pk_c1p = (bf16_t*)carve((size_t)320 * 64 * 2);
  bf16_t* pk_c11 = (bf16_t*)carve((size_t)3 * 320 * 64 * 2);
  bf16_t* pk_c12 = (bf16_t*)carve((size_t)3 * 320 * 320 * 2);
  bf16_t* pk_ihd = (bf16_t*)carve((size_t)1280 * 320 * 2);
  bf16_t* pk_ihp = (bf16_t*)carve((size_t)16 * 1280 * 2);
  bf16_t* pk_wih = (bf16_t*)carve((size_t)192 * 320 * 2);
  bf16_t* pk_whh = (bf16_t*)carve((size_t)192 * 64 * 2);
  bf16_t* pk_out = (bf16_t*)carve((size_t)320 * 64 * 2);
  bf16_t* pk_fcr = (bf16_t*)carve((size_t)16 * 320 * 2);
  float*  mask1  = (float*)carve((size_t)BT * 4);
  float*  y1f    = (float*)carve((size_t)BT * 320 * 4);
  bf16_t* y1b    = (bf16_t*)carve((size_t)BT * 320 * 2);
  bf16_t* y2b    = (bf16_t*)carve((size_t)BT * 320 * 2);
  float*  gi1    = (float*)carve((size_t)BT * 192 * 4);
  float*  gi2    = (float*)carve((size_t)BT * 192 * 4);
  bf16_t* encb   = (bf16_t*)carve((size_t)BT * 320 * 2);
  // transient region A — three non-overlapping-in-time overlays
  char* regA = carve((size_t)176 * 1024 * 1024);
  float*  h0f  = (float*)regA;                                    // BT*64 f32
  bf16_t* gb   = (bf16_t*)(regA + (size_t)BT * 64 * 4);
  bf16_t* g2b  = gb  + (size_t)BT * 64;
  bf16_t* h1b  = g2b + (size_t)BT * 64;
  bf16_t* g3b  = h1b + (size_t)BT * 64;
  float*  res2 = (float*)(g3b + (size_t)BT * 64);                 // BT*320 f32
  bf16_t* g4b  = (bf16_t*)((char*)res2 + (size_t)BT * 320 * 4);   // BT*320 bf16
  bf16_t* g5   = (bf16_t*)regA;                                   // BT*1280 bf16 (interp phase)
  float*  xi   = (float*)(regA + (size_t)BT * 1280 * 2);          // BT*12 f32
  bf16_t* o1b  = (bf16_t*)regA;                                   // BT*320 bf16 (GRU phase)
  bf16_t* o2b  = o1b + (size_t)BT * 320;

  auto blk = [](size_t n) { return dim3((unsigned)((n + 255) / 256)); };
  const int gc64   = (BT / 16) * (64 / 64) / 8;      // 512
  const int gc320  = (BT / 16) * (320 / 64) / 8;     // 2560
  const int gc1280 = (BT / 16) * (1280 / 64) / 8;    // 10240
  const int gc192  = (BT / 16) * (192 / 64) / 8;     // 1536
  const int gc16   = (BT / 16) / 8;                  // 512 (NWT=1)

  // ---- phase 0: weight packing ----
  pack_conv_kernel<<<blk(3 * 64 * 64), 256, 0, stream>>>(c01w, pk_c01, 64, 64);
  pack_conv_kernel<<<blk(3 * 64 * 64), 256, 0, stream>>>(c02w, pk_c02, 64, 64);
  pack_nk_kernel<<<blk(320 * 64), 256, 0, stream>>>(c1pw, pk_c1p, 320, 64, 320, 0);
  pack_conv_kernel<<<blk(3 * 320 * 64), 256, 0, stream>>>(c11w, pk_c11, 320, 64);
  pack_conv_kernel<<<blk(3 * 320 * 320), 256, 0, stream>>>(c12w, pk_c12, 320, 320);
  pack_nk_kernel<<<blk(1280 * 320), 256, 0, stream>>>(ihdw, pk_ihd, 1280, 320, 1280, 1);
  pack_nk_kernel<<<blk(16 * 1280), 256, 0, stream>>>(ihpw, pk_ihp, 16, 1280, 12, 1);
  pack_nk_kernel<<<blk(192 * 320), 256, 0, stream>>>(wih, pk_wih, 192, 320, 192, 0);
  pack_nk_kernel<<<blk(192 * 64), 256, 0, stream>>>(whh, pk_whh, 192, 64, 192, 0);
  pack_nk_kernel<<<blk(320 * 64), 256, 0, stream>>>(outw, pk_out, 320, 64, 320, 1);
  pack_nk_kernel<<<blk(16 * 320), 256, 0, stream>>>(fcrw, pk_fcr, 16, 320, 12, 1);

  // ---- encoder (4 passes) ----
  auto enc_pass = [&](const float* xin, float* encF, bf16_t* encBd) {
    proj_gelu_kernel<<<blk((size_t)BT * 64), 256, 0, stream>>>(xin, w_in, b_in, h0f, gb, BT);
    wconv3_kernel<1, false><<<gc64, 256, 0, stream>>>(gb, pk_c01, c01b, nullptr,
        nullptr, g2b, nullptr, BT, 64, 64, Tc, 1);
    wconv3_kernel<0, true><<<gc64, 256, 0, stream>>>(g2b, pk_c02, c02b, h0f,
        nullptr, h1b, g3b, BT, 64, 64, Tc, 1);
    wgemm_kernel<4, 0, true, false, false><<<gc320, 256, 0, stream>>>(h1b, pk_c1p, c1pb,
        nullptr, res2, nullptr, nullptr, BT, 320, 64, 320, 0, 0);
    wconv3_kernel<1, false><<<gc320, 256, 0, stream>>>(g3b, pk_c11, c11b, nullptr,
        nullptr, g4b, nullptr, BT, 320, 64, Tc, 2);
    wconv3_kernel<0, true><<<gc320, 256, 0, stream>>>(g4b, pk_c12, c12b, res2,
        encF, encBd, nullptr, BT, 320, 320, Tc, 2);
  };
  auto interp_pass = [&](float* yF, bf16_t* yB, float* giOut) {
    wgemm_kernel<4, 2, true, false, false><<<gc1280, 256, 0, stream>>>(encb, pk_ihd, ihdb,
        nullptr, nullptr, g5, nullptr, BT, 1280, 320, 1280, 0, 0);
    wgemm_kernel<1, 0, true, false, false><<<gc16, 256, 0, stream>>>(g5, pk_ihp, ihpb,
        nullptr, xi, nullptr, nullptr, BT, 16, 1280, 12, 0, 0);
    fc_kernel<<<blk((size_t)BT * 320), 256, 0, stream>>>(xi, fcw, fcb, yF, yB, BT);
    wgemm_kernel<4, 0, true, false, true><<<gc192, 256, 0, stream>>>(yB, pk_wih, bih,
        nullptr, giOut, nullptr, nullptr, BT, 192, 320, 192, Tc, Bc);
  };

  enc_pass(X1, out, nullptr);                 // x_whole1 -> d_out
  enc_pass(X2, out + OFF_XW2, nullptr);       // x_whole2 -> d_out
  enc_pass(x1, nullptr, encb);
  interp_pass(y1f, y1b, gi1);
  mask_kernel<<<blk(BT), 256, 0, stream>>>(xi, mask1, BT);   // xi still holds x_interp1
  enc_pass(x2, nullptr, encb);
  interp_pass(nullptr, y2b, gi2);

  // ---- GRU (both chains, 8 row-groups each) ----
  gru_kernel<<<dim3(Bc / 16, 2), 128, 0, stream>>>(gi1, gi2, y1f, mask1,
      pk_whh, bhh, pk_out, outb, o1b, o2b, Tc, Bc);

  // ---- outputs = o @ fcr_w + fcr_b (permute [T,B] -> [B,T]) ----
  wgemm_kernel<1, 0, true, false, true><<<gc16, 256, 0, stream>>>(o1b, pk_fcr, fcrb,
      nullptr, out + OFF_O1, nullptr, nullptr, BT, 16, 320, 12, Bc, Tc);
  wgemm_kernel<1, 0, true, false, true><<<gc16, 256, 0, stream>>>(o2b, pk_fcr, fcrb,
      nullptr, out + OFF_O2, nullptr, nullptr, BT, 16, 320, 12, Bc, Tc);

  // ---- masks + passthrough copies ----
  misc_out_kernel<<<blk((size_t)BT * 12), 256, 0, stream>>>(x1, x2,
      out + OFF_M1, out + OFF_M2, out + OFF_X1C, out + OFF_X2C, BT * 12);
}
